// RouteNet_49460843380928
// MI455X (gfx1250) — matrix operand order repaired and checked by reading, not compiled, not run on hardware
//
#include <hip/hip_runtime.h>
#include <hip/hip_bf16.h>

#define N_LINKS  10000
#define N_PATHS  100000
#define PATH_LEN 8
#define H        32
#define RD       8
#define T_ROUNDS 8

#define NEG_LOG2E  (-1.4426950408889634f)
#define NEG_2LOG2E (-2.8853901617779268f)

typedef __attribute__((ext_vector_type(16))) _Float16 v16h;
typedef __attribute__((ext_vector_type(8)))  _Float16 v8h;
typedef __attribute__((ext_vector_type(8)))  float    v8f;

static __device__ __forceinline__ v8f wmma32(v16h a, v16h b, v8f c) {
  // D = A(16x32 f16) * B(32x16 f16) + C(16x16 f32)
  return __builtin_amdgcn_wmma_f32_16x16x32_f16(false, a, false, b, (short)0, c,
                                                false, false);
}

// sigmoid(x + b), with bs = -log2e * b precomputed: fma + v_exp + add + v_rcp
static __device__ __forceinline__ float sigmoid_b(float x, float bs) {
  float e = __builtin_amdgcn_exp2f(__builtin_fmaf(x, NEG_LOG2E, bs));
  return __builtin_amdgcn_rcpf(1.0f + e);
}
// tanh(y): mul + v_exp + add + v_rcp + sub + mul
static __device__ __forceinline__ float tanh_f(float y) {
  float e = __builtin_amdgcn_exp2f(y * NEG_2LOG2E);
  return (1.0f - e) * __builtin_amdgcn_rcpf(1.0f + e);
}

// ---------------------------------------------------------------------------
// Pack a torch-layout [96, 32] weight matrix into the WMMA B-operand register
// image: for column tile t (6 tiles of 16 gate-columns), lane l holds column
// c = t*16 + (l&15), K = (l>>4)*8 + i + (i>=8 ? 8 : 0)  (i = 0..15), stored f16, 16 halfs/lane.
// mats: 0=p_Wih 1=p_Whh 2=l_Wih 3=l_Whh, each 6*32*16 halfs.
// ---------------------------------------------------------------------------
__global__ void pack_weights(const float* __restrict__ pWih,
                             const float* __restrict__ pWhh,
                             const float* __restrict__ lWih,
                             const float* __restrict__ lWhh,
                             _Float16* __restrict__ wpack) {
  int idx = blockIdx.x * blockDim.x + threadIdx.x;
  if (idx >= 4 * 6 * 32) return;
  int mat  = idx / 192;
  int r    = idx - mat * 192;
  int tile = r >> 5;
  int lane = r & 31;
  const float* W = (mat == 0) ? pWih : (mat == 1) ? pWhh : (mat == 2) ? lWih : lWhh;
  int c   = tile * 16 + (lane & 15);
  int klo = (lane >> 4) * 8;
  _Float16* dst = wpack + mat * (6 * 32 * 16) + (tile * 32 + lane) * 16;
#pragma unroll
  for (int i = 0; i < 16; ++i) dst[i] = (_Float16)W[c * H + klo + i + (i >= 8 ? 8 : 0)];
}

// link_state[:,0]=capacity, path_state[:,0]=bandwidth, rest zero; zero agg.
__global__ void init_states(const float* __restrict__ cap,
                            const float* __restrict__ bw,
                            float* __restrict__ link_state,
                            float* __restrict__ path_state,
                            float* __restrict__ agg) {
  int tid = blockIdx.x * blockDim.x + threadIdx.x;
  const int nl = N_LINKS * H;
  const int np = N_PATHS * H;
  if (tid < nl) {
    link_state[tid] = ((tid & (H - 1)) == 0) ? cap[tid >> 5] : 0.0f;
    agg[tid] = 0.0f;
  } else if (tid < nl + np) {
    int t2 = tid - nl;
    path_state[t2] = ((t2 & (H - 1)) == 0) ? bw[t2 >> 5] : 0.0f;
  }
}

__global__ void zero_agg(float* __restrict__ agg) {
  int tid = blockIdx.x * blockDim.x + threadIdx.x;
  if (tid < N_LINKS * H) agg[tid] = 0.0f;
}

// ---------------------------------------------------------------------------
// Fused per-round path kernel: one wave = one 16-path tile.
// For t = 0..7:  x_t = link_state[links[p*8+t]]   (gather, no rnn_in tensor)
//   gi/gh via v_wmma_f32_16x16x32_f16 (12 WMMAs/step, C-chains seeded with
//   inline 0; biases folded into the gate nonlinearities), gate math in f32,
//   h -> message scatter-add into agg (f32 atomics, L2 resident).
// ---------------------------------------------------------------------------
__global__ void __launch_bounds__(32)
path_gru(const int* __restrict__ links,
         const float* __restrict__ link_state,
         float* __restrict__ path_state,
         float* __restrict__ agg,
         const _Float16* __restrict__ wpack,   // [0]=p_Wih pack, [+3072]=p_Whh
         const float* __restrict__ bih,
         const float* __restrict__ bhh) {
  __shared__ __align__(16) _Float16 hlds[16 * H];   // f16 h tile, row-major 16x32
  __shared__ int lk[16 * PATH_LEN];                 // link ids for tile

  const int lane = threadIdx.x;
  const int base = blockIdx.x * 16;
  const int hi   = lane >> 4;       // row-half selector for C layout
  const int ln   = lane & 15;
  const int kb   = hi ? 8 : 0;      // A-operand K base for this lane

  // Stage the tile's 16x8 link ids: lk[m*8+t] = links[(base+m)*8+t] (contiguous)
#pragma unroll
  for (int i = 0; i < 4; ++i)
    lk[lane * 4 + i] = links[base * PATH_LEN + lane * 4 + i];

  // B operands (register-resident for the whole kernel)
  v16h Bih[6], Bhh[6];
#pragma unroll
  for (int t6 = 0; t6 < 6; ++t6) {
    Bih[t6] = *(const v16h*)(wpack + (t6 * 32 + lane) * 16);
    Bhh[t6] = *(const v16h*)(wpack + 6 * 32 * 16 + (t6 * 32 + lane) * 16);
  }

  // Per-lane bias constants in C layout (gate order r,z,n over 96 columns).
  // r/z biases pre-scaled by -log2e so they fold into the sigmoid fma.
  float brs[2], bzs[2], bni[2], bnh[2];
#pragma unroll
  for (int q = 0; q < 2; ++q) {
    int f  = ln + 16 * q;
    brs[q] = NEG_LOG2E * (bih[f]      + bhh[f]);
    bzs[q] = NEG_LOG2E * (bih[32 + f] + bhh[32 + f]);
    bni[q] = bih[64 + f];
    bnh[q] = bhh[64 + f];
  }

  // h_old in C layout: lane holds 8 rows x 2 features
  float hreg[2][8];
#pragma unroll
  for (int q = 0; q < 2; ++q)
#pragma unroll
    for (int j = 0; j < 8; ++j)
      hreg[q][j] = path_state[(base + j + 8 * hi) * H + ln + 16 * q];

  // seed f16 h tile
#pragma unroll
  for (int q = 0; q < 2; ++q)
#pragma unroll
    for (int j = 0; j < 8; ++j)
      hlds[(j + 8 * hi) * H + ln + 16 * q] = (_Float16)hreg[q][j];
  asm volatile("s_wait_dscnt 0" ::: "memory");

  for (int t = 0; t < PATH_LEN; ++t) {
    // A_x: gather one link row per A-row, convert f32 -> f16 in A layout
    union { v16h v; _Float16 h[16]; } ax;
    {
      int lid = lk[ln * PATH_LEN + t];
      const float* src = link_state + lid * H + kb;
#pragma unroll
      for (int i = 0; i < 8; ++i) {
        ax.h[i]     = (_Float16)src[i];
        ax.h[8 + i] = (_Float16)src[16 + i];
      }
    }
    // A_h from the f16 LDS tile (two b128 DS loads)
    v8h alo = *(const v8h*)&hlds[ln * H + kb];
    v8h ahi = *(const v8h*)&hlds[ln * H + kb + 16];
    v16h ah = __builtin_shufflevector(alo, ahi, 0, 1, 2, 3, 4, 5, 6, 7,
                                      8, 9, 10, 11, 12, 13, 14, 15);

    // WMMA chains seeded from inline constant 0 (SRC2 = 0)
    v8f zero = {};
    v8f cr[2], cz[2], cin[2], chn[2];
#pragma unroll
    for (int q = 0; q < 2; ++q) {
      cr[q]  = wmma32(ax.v, Bih[q],     zero);
      cr[q]  = wmma32(ah,   Bhh[q],     cr[q]);
      cz[q]  = wmma32(ax.v, Bih[2 + q], zero);
      cz[q]  = wmma32(ah,   Bhh[2 + q], cz[q]);
      cin[q] = wmma32(ax.v, Bih[4 + q], zero);   // i_n (kept separate:
      chn[q] = wmma32(ah,   Bhh[4 + q], zero);   //  n = tanh(i_n + r*h_n))
    }

    int lrow[8];
#pragma unroll
    for (int j = 0; j < 8; ++j) lrow[j] = lk[(j + 8 * hi) * PATH_LEN + t];

#pragma unroll
    for (int q = 0; q < 2; ++q) {
      int f = ln + 16 * q;
#pragma unroll
      for (int j = 0; j < 8; ++j) {
        float r = sigmoid_b(cr[q][j], brs[q]);
        float z = sigmoid_b(cz[q][j], bzs[q]);
        float u = __builtin_fmaf(r, chn[q][j], cin[q][j]);
        float v = __builtin_fmaf(r, bnh[q], bni[q]);
        float n = tanh_f(u + v);
        float hnew = __builtin_fmaf(z, hreg[q][j] - n, n);
        hreg[q][j] = hnew;
        hlds[(j + 8 * hi) * H + f] = (_Float16)hnew;
        atomicAdd(&agg[lrow[j] * H + f], hnew);   // segment_sum, fused
      }
    }
    asm volatile("s_wait_dscnt 0" ::: "memory");
  }

#pragma unroll
  for (int q = 0; q < 2; ++q)
#pragma unroll
    for (int j = 0; j < 8; ++j)
      path_state[(base + j + 8 * hi) * H + ln + 16 * q] = hreg[q][j];
}

// ---------------------------------------------------------------------------
// Single-step link GRU: x = agg row, h = link_state. Same WMMA mapping.
// ---------------------------------------------------------------------------
__global__ void __launch_bounds__(32)
link_gru(const float* __restrict__ agg,
         float* __restrict__ link_state,
         const _Float16* __restrict__ wpack,   // [0]=l_Wih pack, [+3072]=l_Whh
         const float* __restrict__ bih,
         const float* __restrict__ bhh) {
  __shared__ __align__(16) _Float16 hlds[16 * H];

  const int lane = threadIdx.x;
  const int base = blockIdx.x * 16;
  const int hi   = lane >> 4;
  const int ln   = lane & 15;
  const int kb   = hi ? 8 : 0;

  v16h Bih[6], Bhh[6];
#pragma unroll
  for (int t6 = 0; t6 < 6; ++t6) {
    Bih[t6] = *(const v16h*)(wpack + (t6 * 32 + lane) * 16);
    Bhh[t6] = *(const v16h*)(wpack + 6 * 32 * 16 + (t6 * 32 + lane) * 16);
  }

  float brs[2], bzs[2], bni[2], bnh[2];
#pragma unroll
  for (int q = 0; q < 2; ++q) {
    int f  = ln + 16 * q;
    brs[q] = NEG_LOG2E * (bih[f]      + bhh[f]);
    bzs[q] = NEG_LOG2E * (bih[32 + f] + bhh[32 + f]);
    bni[q] = bih[64 + f];
    bnh[q] = bhh[64 + f];
  }

  float hreg[2][8];
#pragma unroll
  for (int q = 0; q < 2; ++q)
#pragma unroll
    for (int j = 0; j < 8; ++j)
      hreg[q][j] = link_state[(base + j + 8 * hi) * H + ln + 16 * q];

#pragma unroll
  for (int q = 0; q < 2; ++q)
#pragma unroll
    for (int j = 0; j < 8; ++j)
      hlds[(j + 8 * hi) * H + ln + 16 * q] = (_Float16)hreg[q][j];
  asm volatile("s_wait_dscnt 0" ::: "memory");

  union { v16h v; _Float16 h[16]; } ax;
  {
    const float* src = agg + (base + ln) * H + kb;
#pragma unroll
    for (int i = 0; i < 8; ++i) {
      ax.h[i]     = (_Float16)src[i];
      ax.h[8 + i] = (_Float16)src[16 + i];
    }
  }
  v8h alo = *(const v8h*)&hlds[ln * H + kb];
  v8h ahi = *(const v8h*)&hlds[ln * H + kb + 16];
  v16h ah = __builtin_shufflevector(alo, ahi, 0, 1, 2, 3, 4, 5, 6, 7,
                                    8, 9, 10, 11, 12, 13, 14, 15);

  v8f zero = {};
  v8f cr[2], cz[2], cin[2], chn[2];
#pragma unroll
  for (int q = 0; q < 2; ++q) {
    cr[q]  = wmma32(ax.v, Bih[q],     zero);
    cr[q]  = wmma32(ah,   Bhh[q],     cr[q]);
    cz[q]  = wmma32(ax.v, Bih[2 + q], zero);
    cz[q]  = wmma32(ah,   Bhh[2 + q], cz[q]);
    cin[q] = wmma32(ax.v, Bih[4 + q], zero);
    chn[q] = wmma32(ah,   Bhh[4 + q], zero);
  }

#pragma unroll
  for (int q = 0; q < 2; ++q) {
#pragma unroll
    for (int j = 0; j < 8; ++j) {
      float r = sigmoid_b(cr[q][j], brs[q]);
      float z = sigmoid_b(cz[q][j], bzs[q]);
      float u = __builtin_fmaf(r, chn[q][j], cin[q][j]);
      float v = __builtin_fmaf(r, bnh[q], bni[q]);
      float n = tanh_f(u + v);
      link_state[(base + j + 8 * hi) * H + ln + 16 * q] =
          __builtin_fmaf(z, hreg[q][j] - n, n);
    }
  }
}

// Readout MLP (tiny: 33M MAC total) — plain VALU.
__global__ void readout(const float* __restrict__ path_state,
                        const float* __restrict__ W1, const float* __restrict__ b1,
                        const float* __restrict__ W2, const float* __restrict__ b2,
                        const float* __restrict__ W3, const float* __restrict__ b3,
                        float* __restrict__ out) {
  int p = blockIdx.x * blockDim.x + threadIdx.x;
  if (p >= N_PATHS) return;
  const float* h = path_state + p * H;
  float x1[RD], x2[RD];
#pragma unroll
  for (int k = 0; k < RD; ++k) {
    float a = b1[k];
#pragma unroll
    for (int i = 0; i < H; ++i) a += W1[k * H + i] * h[i];
    x1[k] = fmaxf(a, 0.0f);
  }
#pragma unroll
  for (int k = 0; k < RD; ++k) {
    float a = b2[k];
#pragma unroll
    for (int i = 0; i < RD; ++i) a += W2[k * RD + i] * x1[i];
    x2[k] = fmaxf(a, 0.0f);
  }
  float a = b3[0];
#pragma unroll
  for (int i = 0; i < RD; ++i) a += W3[i] * x2[i];
  out[p] = a;
}

extern "C" void kernel_launch(void* const* d_in, const int* in_sizes, int n_in,
                              void* d_out, int out_size, void* d_ws, size_t ws_size,
                              hipStream_t stream) {
  const int*   links = (const int*)  d_in[0];
  // d_in[1] = paths, d_in[2] = seqs: implied by e = p*8 + s, not needed.
  const float* cap   = (const float*)d_in[3];
  const float* bw    = (const float*)d_in[4];
  const float* pWih  = (const float*)d_in[5];
  const float* pWhh  = (const float*)d_in[6];
  const float* pbih  = (const float*)d_in[7];
  const float* pbhh  = (const float*)d_in[8];
  const float* lWih  = (const float*)d_in[9];
  const float* lWhh  = (const float*)d_in[10];
  const float* lbih  = (const float*)d_in[11];
  const float* lbhh  = (const float*)d_in[12];
  const float* W1    = (const float*)d_in[13];
  const float* b1    = (const float*)d_in[14];
  const float* W2    = (const float*)d_in[15];
  const float* b2    = (const float*)d_in[16];
  const float* W3    = (const float*)d_in[17];
  const float* b3    = (const float*)d_in[18];
  float* out = (float*)d_out;

  char* ws = (char*)d_ws;
  size_t off = 0;
  auto alloc = [&](size_t bytes) {
    size_t p = off;
    off += (bytes + 255) & ~(size_t)255;
    return p;
  };
  float*    link_state = (float*)(ws + alloc((size_t)N_LINKS * H * 4));
  float*    path_state = (float*)(ws + alloc((size_t)N_PATHS * H * 4));
  float*    agg        = (float*)(ws + alloc((size_t)N_LINKS * H * 4));
  _Float16* wpack      = (_Float16*)(ws + alloc((size_t)4 * 6 * 32 * 16 * 2));

  pack_weights<<<3, 256, 0, stream>>>(pWih, pWhh, lWih, lWhh, wpack);
  const int tot = (N_LINKS + N_PATHS) * H;
  init_states<<<(tot + 255) / 256, 256, 0, stream>>>(cap, bw, link_state,
                                                     path_state, agg);
  for (int r = 0; r < T_ROUNDS; ++r) {
    path_gru<<<N_PATHS / 16, 32, 0, stream>>>(links, link_state, path_state,
                                              agg, wpack, pbih, pbhh);
    link_gru<<<N_LINKS / 16, 32, 0, stream>>>(agg, link_state,
                                              wpack + 2 * (6 * 32 * 16),
                                              lbih, lbhh);
    zero_agg<<<(N_LINKS * H + 255) / 256, 256, 0, stream>>>(agg);
  }
  readout<<<(N_PATHS + 255) / 256, 256, 0, stream>>>(path_state, W1, b1, W2,
                                                     b2, W3, b3, out);
}
